// AdditiveAttn_4947802325395
// MI455X (gfx1250) — matrix operand order repaired and checked by reading, not compiled, not run on hardware
//
#include <hip/hip_runtime.h>
#include <hip/hip_bf16.h>
#include <math.h>

// ---------- types ----------
typedef __attribute__((ext_vector_type(16))) _Float16 v16h;
typedef __attribute__((ext_vector_type(8)))  float    v8f;
typedef __attribute__((ext_vector_type(2)))  float    v2f;

#define HD    128     // H*D
#define INDIM 64
#define NHEAD 8
#define DDIM  16
#define CLAMPV 5.0f

// float atomic max via int/uint ordering trick (valid for all finite floats)
__device__ __forceinline__ void atomicMaxF(float* addr, float val) {
    if (val >= 0.0f) {
        atomicMax((int*)addr, __float_as_int(val));
    } else {
        atomicMin((unsigned int*)addr, __float_as_uint(val));
    }
}

// ---------- kernel 0: init accumulators ----------
__global__ void init_kernel(float* m_ws, float* den_ws, float* oV, float* rA,
                            int n128, int n8) {
    int t = blockIdx.x * 256 + threadIdx.x;
    if (t < n128) { oV[t] = 0.0f; rA[t] = 0.0f; }
    if (t < n8)   { m_ws[t] = -CLAMPV; den_ws[t] = 0.0f; }
}

// ---------- kernel 0b: one-time weight f32 -> f16 repack ----------
// Output layout matches the per-lane B-fragment order:
//   idx = ((tile*2 + ks)*32 + lane)*16 + j
//   value = W[k*ncols + tile*16 + (lane&15)],  k = ks*32 + (j<8 ? (lane>>4)*8 + j : 16 + (lane>>4)*8 + (j-8))
// so a wave's B tile (tile, ks) is one contiguous 32B v16h per lane.
__global__ void wprep_kernel(const float* __restrict__ W, _Float16* __restrict__ Wf,
                             int ncols) {
    int t = blockIdx.x * 256 + threadIdx.x;
    if (t >= INDIM * ncols) return;
    const int j    = t & 15;
    const int lane = (t >> 4) & 31;
    const int ks   = (t >> 9) & 1;
    const int tile = t >> 10;
    const int sel  = lane >> 4;
    const int n    = lane & 15;
    const int k    = ks * 32 + (j < 8 ? sel * 8 + j : 16 + sel * 8 + (j - 8));
    Wf[t] = (_Float16)W[(size_t)k * ncols + tile * 16 + n];
}

// Cooperative A-tile (16 rows x 64 K, fp32) -> LDS f16 fragments.
// LDS layout: (ks*32 + lane)*16 + j  (ks = K-step 0/1), A-frag element order:
//   j<8 : K = ks*32 + sel*8 + j ;  j>=8 : K = ks*32 + 16 + sel*8 + (j-8)
__device__ __forceinline__ void stage_A_tile(const float* __restrict__ src,
                                             int row0, _Float16* lds) {
    for (int t = threadIdx.x; t < 1024; t += 256) {
        const int j    = t & 15;
        const int lane = (t >> 4) & 31;
        const int ks   = t >> 9;
        const int sel  = lane >> 4;
        const int row  = lane & 15;
        const int K    = ks * 32 + (j < 8 ? sel * 8 + j : 16 + sel * 8 + (j - 8));
        lds[t] = (_Float16)src[(size_t)(row0 + row) * INDIM + K];
    }
}

// ---------- kernel 1: node projections Q,K,V  (f16 WMMA, K=64) ----------
// grid = N/16 blocks, 256 threads (8 waves). Wave w owns cols [w*16, w*16+16)
// of each of Q/K/V. A tile staged once via LDS; B tiles are contiguous f16.
__global__ void qkv_kernel(const float* __restrict__ x,
                           const _Float16* __restrict__ Qwf,
                           const _Float16* __restrict__ Kwf,
                           const _Float16* __restrict__ Vwf,
                           const float* __restrict__ Qb, const float* __restrict__ Kb,
                           const float* __restrict__ Vb,
                           float* __restrict__ Qh, float* __restrict__ Kh,
                           float* __restrict__ Vh, int N) {
    __shared__ alignas(32) _Float16 ax[1024];
    const int wave = threadIdx.x >> 5;
    const int lane = threadIdx.x & 31;
    const int row0 = blockIdx.x * 16;

    stage_A_tile(x, row0, ax);
    __syncthreads();
    const v16h a0 = *(const v16h*)&ax[lane * 16];
    const v16h a1 = *(const v16h*)&ax[(32 + lane) * 16];

    const _Float16* Wf[3] = {Qwf, Kwf, Vwf};
    const float*    Bs[3] = {Qb, Kb, Vb};
    float*          Os[3] = {Qh, Kh, Vh};
    const int col0 = wave * 16;
    const int n    = lane & 15;
    const int sel  = lane >> 4;

    for (int t = 0; t < 3; ++t) {
        // tile = wave : base = ((wave*2)*32 + lane)*16
        const _Float16* bp = Wf[t] + ((size_t)(wave * 64) + lane) * 16;
        const v16h b0 = *(const v16h*)bp;          // ks = 0
        const v16h b1 = *(const v16h*)(bp + 512);  // ks = 1
        v8f c = {};
        c = __builtin_amdgcn_wmma_f32_16x16x32_f16(false, a0, false, b0,
                                                   (short)0, c, false, false);
        c = __builtin_amdgcn_wmma_f32_16x16x32_f16(false, a1, false, b1,
                                                   (short)0, c, false, false);
        const float bias = Bs[t][col0 + n];
        float* O = Os[t];
        #pragma unroll
        for (int r = 0; r < 8; ++r) {   // D: VGPR r -> row M = r + 8*sel
            O[(size_t)(row0 + r + 8 * sel) * HD + col0 + n] = c[r] + bias;
        }
    }
}

// ---------- kernel 2: edge GEMM + score + oE + a + segment-max ----------
// grid = Ne/16 blocks, 256 threads. Wave h handles head h: two 16x16 E-tiles
// (Ex1 = E cols h*32+d, Ex2 = cols h*32+16+d), 4 f16 WMMAs, then score/a.
__global__ void edge_score_kernel(const float* __restrict__ edge_attr,
                                  const int* __restrict__ eidx,
                                  const _Float16* __restrict__ Ewf,
                                  const float* __restrict__ Eb,
                                  const float* __restrict__ Aw,
                                  const float* __restrict__ Qh, const float* __restrict__ Kh,
                                  float* __restrict__ oE, float* __restrict__ a_ws,
                                  float* __restrict__ m_ws, int Ne) {
    __shared__ alignas(32) _Float16 ax[1024];
    const int h    = threadIdx.x >> 5;
    const int lane = threadIdx.x & 31;
    const int e0   = blockIdx.x * 16;
    const int n    = lane & 15;
    const int sel  = lane >> 4;

    stage_A_tile(edge_attr, e0, ax);
    __syncthreads();
    const v16h a0 = *(const v16h*)&ax[lane * 16];
    const v16h a1 = *(const v16h*)&ax[(32 + lane) * 16];

    // Ewf tiles: tile1 = 2h (Ex1), tile2 = 2h+1 (Ex2); base = (128h + lane)*16
    const _Float16* bp = Ewf + (size_t)(128 * h + lane) * 16;
    v8f c1 = {}, c2 = {};
    {
        const v16h b10 = *(const v16h*)bp;             // tile1 ks0
        const v16h b11 = *(const v16h*)(bp + 512);     // tile1 ks1
        c1 = __builtin_amdgcn_wmma_f32_16x16x32_f16(false, a0, false, b10,
                                                    (short)0, c1, false, false);
        c1 = __builtin_amdgcn_wmma_f32_16x16x32_f16(false, a1, false, b11,
                                                    (short)0, c1, false, false);
    }
    {
        const v16h b20 = *(const v16h*)(bp + 1024);    // tile2 ks0
        const v16h b21 = *(const v16h*)(bp + 1536);    // tile2 ks1
        c2 = __builtin_amdgcn_wmma_f32_16x16x32_f16(false, a0, false, b20,
                                                    (short)0, c2, false, false);
        c2 = __builtin_amdgcn_wmma_f32_16x16x32_f16(false, a1, false, b21,
                                                    (short)0, c2, false, false);
    }

    const float eb1 = Eb[h * 32 + n];
    const float eb2 = Eb[h * 32 + 16 + n];
    const float aw  = Aw[(size_t)n * NHEAD + h];   // Aw[d, h, 0]

    #pragma unroll
    for (int r = 0; r < 8; ++r) {
        const int e  = e0 + r + 8 * sel;           // D-frag row for this VGPR
        const int s  = eidx[e];
        const int dd = eidx[Ne + e];
        const float s2    = (c1[r] + eb1) * (c2[r] + eb2);
        const float ssqrt = copysignf(sqrtf(fabsf(s2)), s2);
        const float score = Kh[(size_t)s * HD + h * DDIM + n]
                          + Qh[(size_t)dd * HD + h * DDIM + n] + ssqrt;
        oE[(size_t)e * HD + h * DDIM + n] = score;

        // a[e,h] = sum_d score*Aw : reduce over the 16 lanes of this half-wave
        float v = score * aw;
        v += __shfl_xor(v, 1);
        v += __shfl_xor(v, 2);
        v += __shfl_xor(v, 4);
        v += __shfl_xor(v, 8);
        if (n == 0) {
            const float aa = fminf(fmaxf(v, -CLAMPV), CLAMPV);
            a_ws[(size_t)e * NHEAD + h] = aa;
            atomicMaxF(&m_ws[(size_t)dd * NHEAD + h], aa);
        }
    }
}

// ---------- kernel 3: ea = exp(a - m[dst]); denom += ea ----------
__global__ void softmax_prep_kernel(const int* __restrict__ eidx,
                                    float* __restrict__ a_ws,
                                    const float* __restrict__ m_ws,
                                    float* __restrict__ den_ws, int Ne) {
    int t = blockIdx.x * 256 + threadIdx.x;
    if (t >= Ne * NHEAD) return;
    const int e = t >> 3, h = t & 7;
    const int dd = eidx[Ne + e];
    const float ea = expf(a_ws[t] - m_ws[(size_t)dd * NHEAD + h]);
    a_ws[t] = ea;   // reuse slot: now holds ea
    atomicAdd(&den_ws[(size_t)dd * NHEAD + h], ea);
}

// ---------- kernel 4: attn scatter into oV / rowAcc ----------
__global__ void scatter_kernel(const int* __restrict__ eidx,
                               const float* __restrict__ ea_ws,
                               const float* __restrict__ den_ws,
                               const float* __restrict__ oE,
                               const float* __restrict__ Vh,
                               float* __restrict__ oV, float* __restrict__ rA, int Ne) {
    long t = (long)blockIdx.x * 256 + threadIdx.x;
    if (t >= (long)Ne * HD) return;
    const int e = (int)(t >> 7);
    const int c = (int)(t & 127);
    const int h = c >> 4;
    const int s  = eidx[e];
    const int dd = eidx[Ne + e];
    const float attn = ea_ws[(size_t)e * NHEAD + h]
                     / (den_ws[(size_t)dd * NHEAD + h] + 1e-16f);
    atomicAdd(&oV[(size_t)dd * HD + c], Vh[(size_t)s * HD + c] * attn);
    atomicAdd(&rA[(size_t)dd * HD + c], oE[(size_t)t] * attn);
}

// ---------- kernel 5: n_out = oV + rowAcc @ VeRow (per head, f32 WMMA K=4) ----------
// grid = N/16 blocks, 256 threads; wave h does head h's 16x16x16 matmul.
__global__ void nodeout_kernel(const float* __restrict__ rA,
                               const float* __restrict__ oV,
                               const float* __restrict__ VeRow,
                               float* __restrict__ n_out, int N) {
    const int h    = threadIdx.x >> 5;
    const int lane = threadIdx.x & 31;
    const int n0   = blockIdx.x * 16;
    const int mrow = lane & 15;
    const int sel  = lane >> 4;

    v8f c = {};
    #pragma unroll
    for (int kk = 0; kk < 4; ++kk) {
        const int k0 = kk * 4 + sel * 2;   // A: lanes0-15 K={k0,k0+1}; 16-31 K={k0+2,k0+3}
        v2f a, b;
        a.x = rA[(size_t)(n0 + mrow) * HD + h * DDIM + k0];
        a.y = rA[(size_t)(n0 + mrow) * HD + h * DDIM + k0 + 1];
        b.x = VeRow[(size_t)(k0)     * HD + h * DDIM + mrow];   // VeRow[k, h, n]
        b.y = VeRow[(size_t)(k0 + 1) * HD + h * DDIM + mrow];
        c = __builtin_amdgcn_wmma_f32_16x16x4_f32(false, a, false, b,
                                                  (short)0, c, false, false);
    }
    #pragma unroll
    for (int r = 0; r < 8; ++r) {
        const size_t idx = (size_t)(n0 + r + 8 * sel) * HD + h * DDIM + mrow;
        n_out[idx] = oV[idx] + c[r];
    }
}

// ---------- host launcher ----------
extern "C" void kernel_launch(void* const* d_in, const int* in_sizes, int n_in,
                              void* d_out, int out_size, void* d_ws, size_t ws_size,
                              hipStream_t stream) {
    const float* x         = (const float*)d_in[0];
    const float* edge_attr = (const float*)d_in[1];
    const int*   eidx      = (const int*)  d_in[2];
    const float* Qw = (const float*)d_in[3];  const float* Qb = (const float*)d_in[4];
    const float* Kw = (const float*)d_in[5];  const float* Kb = (const float*)d_in[6];
    const float* Ew = (const float*)d_in[7];  const float* Eb = (const float*)d_in[8];
    const float* Vw = (const float*)d_in[9];  const float* Vb = (const float*)d_in[10];
    const float* Aw = (const float*)d_in[11];
    const float* VeRow = (const float*)d_in[12];

    const int N  = in_sizes[0] / INDIM;   // 20000
    const int Ne = in_sizes[1] / INDIM;   // 320000

    float* out   = (float*)d_out;
    float* n_out = out;                         // N * 128
    float* oE    = out + (size_t)N * HD;        // Ne * 128

    float* ws   = (float*)d_ws;
    float* Qh   = ws;
    float* Kh   = Qh   + (size_t)N  * HD;
    float* Vh   = Kh   + (size_t)N  * HD;
    float* a_ws = Vh   + (size_t)N  * HD;       // Ne * 8 (a, then ea)
    float* m_ws = a_ws + (size_t)Ne * NHEAD;    // N * 8
    float* den  = m_ws + (size_t)N  * NHEAD;    // N * 8
    float* oV   = den  + (size_t)N  * NHEAD;    // N * 128
    float* rA   = oV   + (size_t)N  * HD;       // N * 128
    // f16 repacked weights (32B-aligned: preceding float count is even)
    _Float16* Qwf = (_Float16*)(rA + (size_t)N * HD);
    _Float16* Kwf = Qwf + (size_t)INDIM * HD;   // 8192 halves each
    _Float16* Vwf = Kwf + (size_t)INDIM * HD;
    _Float16* Ewf = Vwf + (size_t)INDIM * HD;   // 16384 halves

    const int n128 = N * HD, n8 = N * NHEAD;
    init_kernel<<<(n128 + 255) / 256, 256, 0, stream>>>(m_ws, den, oV, rA, n128, n8);

    wprep_kernel<<<(INDIM * HD  + 255) / 256, 256, 0, stream>>>(Qw, Qwf, HD);
    wprep_kernel<<<(INDIM * HD  + 255) / 256, 256, 0, stream>>>(Kw, Kwf, HD);
    wprep_kernel<<<(INDIM * HD  + 255) / 256, 256, 0, stream>>>(Vw, Vwf, HD);
    wprep_kernel<<<(INDIM * 256 + 255) / 256, 256, 0, stream>>>(Ew, Ewf, 256);

    qkv_kernel<<<N / 16, 256, 0, stream>>>(x, Qwf, Kwf, Vwf, Qb, Kb, Vb,
                                           Qh, Kh, Vh, N);

    edge_score_kernel<<<Ne / 16, 256, 0, stream>>>(edge_attr, eidx, Ewf, Eb, Aw,
                                                   Qh, Kh, oE, a_ws, m_ws, Ne);

    softmax_prep_kernel<<<(Ne * NHEAD + 255) / 256, 256, 0, stream>>>(eidx, a_ws, m_ws, den, Ne);

    const long scatterN = (long)Ne * HD;
    scatter_kernel<<<(int)((scatterN + 255) / 256), 256, 0, stream>>>(eidx, a_ws, den,
                                                                     oE, Vh, oV, rA, Ne);

    nodeout_kernel<<<N / 16, 256, 0, stream>>>(rA, oV, VeRow, n_out, N);
}
